// EFNet_att_track_freq_fusion_20822001451362
// MI455X (gfx1250) — compile-verified, matrix operand order checked by reading, not run on hardware
//
#include <hip/hip_runtime.h>
#include <stdint.h>

// ---------------------------------------------------------------------------
// EFNet attention / frequency fusion for MI455X (gfx1250, wave32, WMMA).
// All 1x1 convs + MHA projections run on v_wmma_f32_16x16x32_f16; conv weights
// are staged into LDS via async global->LDS copies (ASYNCcnt) when available.
// Workspace requirement: 8 * 33.55MB tensor slots + 192KB f16 weights ~ 269MB.
// ---------------------------------------------------------------------------

typedef _Float16 half_t;
typedef __attribute__((ext_vector_type(16))) _Float16 v16h;
typedef __attribute__((ext_vector_type(8)))  _Float16 v8h;
typedef __attribute__((ext_vector_type(8)))  float    v8f;
typedef __attribute__((ext_vector_type(4)))  int      v4i;

#define CB   8
#define CC   64
#define CH   128
#define CW   128
#define CHW  16384            // H*W
#define NTOT 8388608          // B*C*H*W
#define NPIX 131072           // B*H*W

#if __has_builtin(__builtin_amdgcn_global_load_async_to_lds_b128) && \
    __has_builtin(__builtin_amdgcn_s_wait_asynccnt)
#define HAS_ASYNC_LDS 1
#else
#define HAS_ASYNC_LDS 0
#endif

// 16-byte global -> LDS copy; async (ASYNCcnt) when the builtin exists.
// Builtin signature (from hipcc diagnostic): (int4* as1, int4* as3, Ii, Ii).
__device__ __forceinline__ void cp16_g2l(const void* g, void* l) {
#if HAS_ASYNC_LDS
  __attribute__((address_space(1))) v4i* gp =
      (__attribute__((address_space(1))) v4i*)(uintptr_t)g;
  __attribute__((address_space(3))) v4i* lp =
      (__attribute__((address_space(3))) v4i*)(uint32_t)(uintptr_t)l;
  __builtin_amdgcn_global_load_async_to_lds_b128(gp, lp, 0, 0);
#else
  *(uint4*)l = *(const uint4*)g;
#endif
}
__device__ __forceinline__ void cp_wait_all() {
#if HAS_ASYNC_LDS
  __builtin_amdgcn_s_wait_asynccnt(0);
#endif
  __syncthreads();
}

// ------------------------------ LayerNorm ----------------------------------
// nn.LayerNorm([C,H,W]) fresh each call: weight=1, bias=0, eps=1e-5.
__global__ void __launch_bounds__(1024)
k_layernorm(const float* __restrict__ x, float* __restrict__ o) {
  __shared__ float ssum[1024];
  __shared__ float ssq[1024];
  const int b = blockIdx.x;
  const int t = threadIdx.x;
  const float* xb = x + (size_t)b * (CC * CHW);
  float s = 0.f, q = 0.f;
  for (int i = t; i < CC * CHW; i += 1024) { float v = xb[i]; s += v; q += v * v; }
  ssum[t] = s; ssq[t] = q;
  __syncthreads();
  for (int off = 512; off; off >>= 1) {
    if (t < off) { ssum[t] += ssum[t + off]; ssq[t] += ssq[t + off]; }
    __syncthreads();
  }
  const float inv_n = 1.0f / (float)(CC * CHW);
  const float mean = ssum[0] * inv_n;
  const float var  = ssq[0] * inv_n - mean * mean;
  const float rstd = rsqrtf(var + 1e-5f);
  float* ob = o + (size_t)b * (CC * CHW);
  for (int i = t; i < CC * CHW; i += 1024) ob[i] = (xb[i] - mean) * rstd;
}

// ------------------------- 1x1 conv via WMMA -------------------------------
// out[b,n,hw] = act( bias[n] + sum_k W[n,k] * in(k)[b,:,hw] ),  n in [0,64)
// in(k): k<64 -> in0 channel k, else in1 channel k-64 (channel-concat input).
// Weight matrix (64 x cin f16, <=16KB) is staged once per block into LDS via
// async global->LDS b128 copies; B fragments then come from ds_load_b128.
// One wave computes a 16-pixel x 64-output tile; M tiles = NPIX/16 = 8192.
// act: 0=none 1=relu 2=sigmoid
__global__ void __launch_bounds__(256)
k_conv1x1_wmma(const float* __restrict__ in0, const float* __restrict__ in1,
               const half_t* __restrict__ w, const float* __restrict__ bias,
               float* __restrict__ out, int cin, int act) {
  __shared__ __align__(16) half_t sw[64 * 128];   // up to 64 x 128 f16 = 16KB

  // Cooperative weight staging: 256 threads x 16B per round.
  {
    const int nbytes = 64 * cin * (int)sizeof(half_t);
    const char* gsrc = (const char*)w;
    char* ldst = (char*)sw;
    for (int off = threadIdx.x * 16; off < nbytes; off += 256 * 16)
      cp16_g2l(gsrc + off, ldst + off);
    cp_wait_all();
  }

  const int lane = threadIdx.x & 31;
  const int wid  = threadIdx.x >> 5;
  const int tile = blockIdx.x * 8 + wid;
  const int p0   = tile << 4;          // first pixel of tile (16-aligned)
  const int b    = p0 >> 14;           // CHW = 16384
  const int hw0  = p0 & (CHW - 1);
  const int col  = lane & 15;          // A row / B,C,D column within tile
  const int hi   = lane >> 4;          // lane-half selector for K striping

  v8f acc[4];
#pragma unroll
  for (int t = 0; t < 4; ++t) {
    const float bv = bias[t * 16 + col];
#pragma unroll
    for (int r = 0; r < 8; ++r) acc[t][r] = bv;
  }

  const int hwA = hw0 + col;           // pixel feeding this lane's A row
  for (int k0 = 0; k0 < cin; k0 += 32) {
    const int kb = k0 + hi * 8;
    // A fragment: 16x32 f16, ISA lane layout (halves 0..7 -> K kb..kb+7,
    // halves 8..15 -> K kb+16..kb+23).
    v16h a;
#pragma unroll
    for (int j = 0; j < 8; ++j) {
      const int k1 = kb + j, k2 = kb + 16 + j;
      const float* s1 = (k1 < CC) ? in0 : in1;
      const float* s2 = (k2 < CC) ? in0 : in1;
      a[j]     = (half_t)s1[((b << 6) + (k1 & (CC - 1))) * CHW + hwA];
      a[8 + j] = (half_t)s2[((b << 6) + (k2 & (CC - 1))) * CHW + hwA];
    }
    if (k0 + 32 < cin) {  // prefetch next K slice of activations
      const int kn = k0 + 32 + hi * 8;
      const float* sp = (kn < CC) ? in0 : in1;
      __builtin_prefetch(&sp[((b << 6) + (kn & (CC - 1))) * CHW + hwA], 0, 1);
    }
#pragma unroll
    for (int t = 0; t < 4; ++t) {
      // B fragment rows live in LDS; the two 8-half K chunks are 16B aligned.
      const half_t* wr = sw + (t * 16 + col) * cin;
      const v8h lo = *(const v8h*)(wr + kb);
      const v8h hs = *(const v8h*)(wr + kb + 16);
      v16h bf;
#pragma unroll
      for (int j = 0; j < 8; ++j) { bf[j] = lo[j]; bf[8 + j] = hs[j]; }
      acc[t] = __builtin_amdgcn_wmma_f32_16x16x32_f16(
          false, a, false, bf, (short)0, acc[t], false, false);
    }
  }

#pragma unroll
  for (int t = 0; t < 4; ++t) {
    float* op = out + ((size_t)((b << 6) + t * 16 + col)) * CHW + hw0;
#pragma unroll
    for (int r = 0; r < 8; ++r) {
      const int m = hi * 8 + r;       // D layout: VGPR r -> row r / 8+r
      float v = acc[t][r];
      if (act == 1) v = fmaxf(v, 0.f);
      else if (act == 2) v = 1.f / (1.f + __expf(-v));
      op[m] = v;
    }
  }
}

// ------------------------- depthwise 3x3, pad 1 ----------------------------
// act: 0=none 3=exact gelu
__global__ void __launch_bounds__(256)
k_dwconv3(const float* __restrict__ x, const float* __restrict__ wt,
          const float* __restrict__ bias, float* __restrict__ o, int act) {
  const int idx = blockIdx.x * 256 + threadIdx.x;
  if (idx >= NTOT) return;
  const int wq = idx & (CW - 1);
  const int h  = (idx >> 7) & (CH - 1);
  const int c  = (idx >> 14) & (CC - 1);
  const int b  = idx >> 20;
  const float* xp = x + ((size_t)((b << 6) + c)) * CHW;
  const float* wp = wt + c * 9;
  float s = bias[c];
#pragma unroll
  for (int kh = 0; kh < 3; ++kh) {
    const int hh = h + kh - 1;
    if (hh < 0 || hh >= CH) continue;
#pragma unroll
    for (int kw = 0; kw < 3; ++kw) {
      const int ww = wq + kw - 1;
      if (ww < 0 || ww >= CW) continue;
      s += xp[hh * CW + ww] * wp[kh * 3 + kw];
    }
  }
  if (act == 3) s = 0.5f * s * (1.f + erff(s * 0.70710678118654752f));
  o[idx] = s;
}

// ------------------- 128-point radix-2 FFT (in place) ----------------------
// One block per transform. inner=1: along W; inner=CW: along H (stride CW).
// dir=-1 forward; dir=+1 inverse with 1/128 scaling (numpy convention).
__global__ void __launch_bounds__(128)
k_fft128(float* __restrict__ re, float* __restrict__ im, int inner, int dir) {
  __shared__ float sr[128];
  __shared__ float si[128];
  const int i = blockIdx.x;
  const size_t base = (size_t)(i / inner) * (128 * inner) + (i % inner);
  const int t = threadIdx.x;
  const int rv = (int)(__brev((unsigned)t) >> 25);  // 7-bit reverse
  sr[rv] = re[base + (size_t)t * inner];
  si[rv] = im[base + (size_t)t * inner];
  __syncthreads();
  for (int len = 2; len <= 128; len <<= 1) {
    const int half = len >> 1;
    if (t < 64) {
      const int grp = t / half, pos = t % half;
      const int i0 = grp * len + pos, i1 = i0 + half;
      const float ang = (float)dir * 6.2831853071795864f * (float)pos / (float)len;
      const float c = cosf(ang), s = sinf(ang);
      const float xr = sr[i1] * c - si[i1] * s;
      const float xi = sr[i1] * s + si[i1] * c;
      const float ur = sr[i0], ui = si[i0];
      sr[i0] = ur + xr; si[i0] = ui + xi;
      sr[i1] = ur - xr; si[i1] = ui - xi;
    }
    __syncthreads();
  }
  const float scale = (dir > 0) ? (1.0f / 128.0f) : 1.0f;
  re[base + (size_t)t * inner] = sr[t] * scale;
  im[base + (size_t)t * inner] = si[t] * scale;
}

// ------------------ per-pixel MHA core (heads=1, L=S=8) --------------------
// scores[l,s] = (1/8) * sum_c qp[l,c,n]*kp[s,c,n]; softmax over s; o = attn@vp
__global__ void __launch_bounds__(256)
k_mha(const float* __restrict__ qp, const float* __restrict__ kp,
      const float* __restrict__ vp, float* __restrict__ o) {
  const int lane = threadIdx.x & 31;
  const int wid  = threadIdx.x >> 5;
  const int n    = blockIdx.x * 8 + wid;        // pixel index in [0, CHW)
  float q[8][2], k[8][2], v[8][2];
#pragma unroll
  for (int l = 0; l < 8; ++l)
#pragma unroll
    for (int j = 0; j < 2; ++j) {
      const size_t idx = ((size_t)(l * CC + lane + 32 * j)) * CHW + n;
      q[l][j] = qp[idx]; k[l][j] = kp[idx]; v[l][j] = vp[idx];
    }
  float attn[8][8];
#pragma unroll
  for (int l = 0; l < 8; ++l)
#pragma unroll
    for (int s = 0; s < 8; ++s) {
      float p = q[l][0] * k[s][0] + q[l][1] * k[s][1];
#pragma unroll
      for (int off = 16; off; off >>= 1) p += __shfl_xor(p, off, 32);
      attn[l][s] = p * 0.125f;                  // 1/sqrt(64)
    }
#pragma unroll
  for (int l = 0; l < 8; ++l) {
    float mx = attn[l][0];
#pragma unroll
    for (int s = 1; s < 8; ++s) mx = fmaxf(mx, attn[l][s]);
    float sum = 0.f;
#pragma unroll
    for (int s = 0; s < 8; ++s) { attn[l][s] = __expf(attn[l][s] - mx); sum += attn[l][s]; }
    const float inv = 1.f / sum;
#pragma unroll
    for (int s = 0; s < 8; ++s) attn[l][s] *= inv;
  }
#pragma unroll
  for (int l = 0; l < 8; ++l)
#pragma unroll
    for (int j = 0; j < 2; ++j) {
      float acc = 0.f;
#pragma unroll
      for (int s = 0; s < 8; ++s) acc += attn[l][s] * v[s][j];
      o[((size_t)(l * CC + lane + 32 * j)) * CHW + n] = acc;
    }
}

// ----------------------------- elementwise ---------------------------------
__global__ void k_zero(float* a, int n) {
  int i = blockIdx.x * 256 + threadIdx.x; if (i < n) a[i] = 0.f;
}
__global__ void k_copy(const float* a, float* o, int n) {
  int i = blockIdx.x * 256 + threadIdx.x; if (i < n) o[i] = a[i];
}
__global__ void k_mul(const float* a, const float* b, float* o, int n) {
  int i = blockIdx.x * 256 + threadIdx.x; if (i < n) o[i] = a[i] * b[i];
}
__global__ void k_add(const float* a, const float* b, float* o, int n) {
  int i = blockIdx.x * 256 + threadIdx.x; if (i < n) o[i] = a[i] + b[i];
}
__global__ void k_add3(const float* a, const float* b, const float* c, float* o, int n) {
  int i = blockIdx.x * 256 + threadIdx.x; if (i < n) o[i] = a[i] + b[i] + c[i];
}
__global__ void k_mag(const float* a, const float* b, float* o, int n) {
  int i = blockIdx.x * 256 + threadIdx.x;
  if (i < n) o[i] = sqrtf(a[i] * a[i] + b[i] * b[i]);
}
__global__ void k_cvt16(const float* s, half_t* d, int n) {
  int i = blockIdx.x * 256 + threadIdx.x; if (i < n) d[i] = (half_t)s[i];
}

// ------------------------------- driver ------------------------------------
extern "C" void kernel_launch(void* const* d_in, const int* in_sizes, int n_in,
                              void* d_out, int out_size, void* d_ws, size_t ws_size,
                              hipStream_t stream) {
  (void)in_sizes; (void)n_in; (void)out_size; (void)ws_size;
  // d_in order (setup_inputs dict insertion order, params depth-first):
  const float* EV = (const float*)d_in[0];
  const float* IM = (const float*)d_in[1];
  const float* c1_b = (const float*)d_in[3];
  const float* dw1_w = (const float*)d_in[4];  const float* dw1_b = (const float*)d_in[5];
  const float* dw2_w = (const float*)d_in[6];  const float* dw2_b = (const float*)d_in[7];
  const float* dw3_w = (const float*)d_in[8];  const float* dw3_b = (const float*)d_in[9];
  const float* cr_b  = (const float*)d_in[11];
  const float* ci_b  = (const float*)d_in[13];
  const float* c2_b  = (const float*)d_in[15];
  const float* c3r_b = (const float*)d_in[17];
  const float* c3i_b = (const float*)d_in[19];
  const float* c4r_b = (const float*)d_in[21];
  const float* c4i_b = (const float*)d_in[23];
  const float* c6_b  = (const float*)d_in[25];
  const float* c7_b  = (const float*)d_in[27];
  const float* ca_b  = (const float*)d_in[29];
  const float* fg_b  = (const float*)d_in[31];
  const float* c5_b  = (const float*)d_in[33];
  const float* mr_in_b  = (const float*)d_in[35];
  const float* mr_out_b = (const float*)d_in[37];
  const float* mi_in_b  = (const float*)d_in[39];
  const float* mi_out_b = (const float*)d_in[41];

  // Workspace: 8 tensor slots + f16 weight staging.
  float* W0 = (float*)d_ws;
  float* buf[9];
  for (int i = 1; i <= 8; ++i) buf[i] = W0 + (size_t)(i - 1) * NTOT;
  half_t* wb = (half_t*)(W0 + (size_t)8 * NTOT);

  // f16 weight staging table: {d_in index, half offset, count}
  const struct { int idx; int off; int n; } WL[] = {
    {2,     0, 8192},  // c1  [64,128]
    {32, 8192, 8192},  // c5  [64,128]
    {10, 16384, 4096}, {12, 20480, 4096}, {14, 24576, 4096},  // cr ci c2
    {16, 28672, 4096}, {18, 32768, 4096},                      // c3r c3i
    {20, 36864, 4096}, {22, 40960, 4096},                      // c4r c4i
    {24, 45056, 4096}, {26, 49152, 4096},                      // c6 c7
    {28, 53248, 4096}, {30, 57344, 4096},                      // ca fg
    {34, 61440, 12288},                                        // mr_in [192,64]
    {36, 73728, 4096},                                         // mr_out
    {38, 77824, 12288},                                        // mi_in
    {40, 90112, 4096},                                         // mi_out
  };
  for (const auto& e : WL)
    k_cvt16<<<(e.n + 255) / 256, 256, 0, stream>>>((const float*)d_in[e.idx], wb + e.off, e.n);

  const half_t *w_c1 = wb, *w_c5 = wb + 8192, *w_cr = wb + 16384, *w_ci = wb + 20480,
               *w_c2 = wb + 24576, *w_c3r = wb + 28672, *w_c3i = wb + 32768,
               *w_c4r = wb + 36864, *w_c4i = wb + 40960, *w_c6 = wb + 45056,
               *w_c7 = wb + 49152, *w_ca = wb + 53248, *w_fg = wb + 57344,
               *w_mr_in = wb + 61440, *w_mr_out = wb + 73728,
               *w_mi_in = wb + 77824, *w_mi_out = wb + 90112;

  const int EWB = (NTOT + 255) / 256;     // 32768 elementwise blocks
  const int CVB = 1024;                   // conv blocks: 8192 waves / 8
  const int FTB = CB * CC * CH;           // 65536 length-128 transforms

  auto conv = [&](const float* i0, const float* i1, const half_t* w,
                  const float* bi, float* o, int cin, int act) {
    k_conv1x1_wmma<<<CVB, 256, 0, stream>>>(i0, i1 ? i1 : i0, w, bi, o, cin, act);
  };

  // x1 = LN(ev), x2 = LN(im)
  k_layernorm<<<CB, 1024, 0, stream>>>(EV, buf[1]);
  k_layernorm<<<CB, 1024, 0, stream>>>(IM, buf[2]);
  // x3 = c1(concat(x1,x2))
  conv(buf[1], buf[2], w_c1, c1_b, buf[3], 128, 0);
  // x4 = fft2(dw1(x3))
  k_dwconv3<<<EWB, 256, 0, stream>>>(buf[3], dw1_w, dw1_b, buf[4], 0);
  k_zero<<<EWB, 256, 0, stream>>>(buf[5], NTOT);
  k_fft128<<<FTB, 128, 0, stream>>>(buf[4], buf[5], 1,  -1);   // rows (W)
  k_fft128<<<FTB, 128, 0, stream>>>(buf[4], buf[5], CW, -1);   // cols (H)
  // real/imag branches -> x5 gate
  conv(buf[4], nullptr, w_cr, cr_b, buf[6], CC, 1);
  conv(buf[5], nullptr, w_ci, ci_b, buf[7], CC, 1);
  conv(buf[6], buf[7], w_c5, c5_b, buf[8], 128, 2);            // x5 = sigmoid
  // x6 = fft2(dw2(c2(x2)))
  conv(buf[2], nullptr, w_c2, c2_b, buf[6], CC, 0);
  k_dwconv3<<<EWB, 256, 0, stream>>>(buf[6], dw2_w, dw2_b, buf[7], 0);
  k_zero<<<EWB, 256, 0, stream>>>(buf[6], NTOT);
  k_fft128<<<FTB, 128, 0, stream>>>(buf[7], buf[6], 1,  -1);
  k_fft128<<<FTB, 128, 0, stream>>>(buf[7], buf[6], CW, -1);
  // prod = x5 * x6
  k_mul<<<EWB, 256, 0, stream>>>(buf[7], buf[8], buf[7], NTOT);
  k_mul<<<EWB, 256, 0, stream>>>(buf[6], buf[8], buf[6], NTOT);
  // pr = c3r(relu(c4r(prod.re))), pi = c3i(relu(c4i(prod.im)))
  conv(buf[7], nullptr, w_c4r, c4r_b, buf[8], CC, 1);
  conv(buf[8], nullptr, w_c3r, c3r_b, buf[7], CC, 0);
  conv(buf[6], nullptr, w_c4i, c4i_b, buf[8], CC, 1);
  conv(buf[8], nullptr, w_c3i, c3i_b, buf[6], CC, 0);
  // x7 = ifft2(pr + i*pi)
  k_fft128<<<FTB, 128, 0, stream>>>(buf[7], buf[6], 1,  +1);
  k_fft128<<<FTB, 128, 0, stream>>>(buf[7], buf[6], CW, +1);
  // x8 = fft(x7, axis=-1)
  k_copy<<<EWB, 256, 0, stream>>>(buf[7], buf[4], NTOT);
  k_copy<<<EWB, 256, 0, stream>>>(buf[6], buf[5], NTOT);
  k_fft128<<<FTB, 128, 0, stream>>>(buf[4], buf[5], 1, -1);
  // x9 = fg(ca(x7.re + x3 + x7.im))
  k_add3<<<EWB, 256, 0, stream>>>(buf[7], buf[3], buf[6], buf[8], NTOT);
  conv(buf[8], nullptr, w_ca, ca_b, buf[3], CC, 0);
  conv(buf[3], nullptr, w_fg, fg_b, buf[8], CC, 0);
  // x10 = ifft(x8 * x9, axis=-1)
  k_mul<<<EWB, 256, 0, stream>>>(buf[4], buf[8], buf[4], NTOT);
  k_mul<<<EWB, 256, 0, stream>>>(buf[5], buf[8], buf[5], NTOT);
  k_fft128<<<FTB, 128, 0, stream>>>(buf[4], buf[5], 1, +1);
  // kv = c6(x1)
  conv(buf[1], nullptr, w_c6, c6_b, buf[6], CC, 0);
  // MHA real: qp/kp/vp -> attention -> out-proj
  conv(buf[4], nullptr, w_mr_in,            mr_in_b,       buf[1], CC, 0);  // Q
  conv(buf[6], nullptr, w_mr_in + 64 * CC,  mr_in_b + 64,  buf[3], CC, 0);  // K
  conv(buf[6], nullptr, w_mr_in + 128 * CC, mr_in_b + 128, buf[7], CC, 0);  // V
  k_mha<<<CHW / 8, 256, 0, stream>>>(buf[1], buf[3], buf[7], buf[8]);
  conv(buf[8], nullptr, w_mr_out, mr_out_b, buf[1], CC, 0);                 // xr
  // MHA imag
  conv(buf[5], nullptr, w_mi_in,            mi_in_b,       buf[4], CC, 0);
  conv(buf[6], nullptr, w_mi_in + 64 * CC,  mi_in_b + 64,  buf[3], CC, 0);
  conv(buf[6], nullptr, w_mi_in + 128 * CC, mi_in_b + 128, buf[7], CC, 0);
  k_mha<<<CHW / 8, 256, 0, stream>>>(buf[4], buf[3], buf[7], buf[8]);
  conv(buf[8], nullptr, w_mi_out, mi_out_b, buf[4], CC, 0);                 // xi
  // mag -> dw3+gelu -> c7 -> + x2
  k_mag<<<EWB, 256, 0, stream>>>(buf[1], buf[4], buf[5], NTOT);
  k_dwconv3<<<EWB, 256, 0, stream>>>(buf[5], dw3_w, dw3_b, buf[6], 3);
  conv(buf[6], nullptr, w_c7, c7_b, buf[7], CC, 0);
  k_add<<<EWB, 256, 0, stream>>>(buf[7], buf[2], (float*)d_out, NTOT);
}
